// GL_TAGConv_3l_512h_w_k3_gnorm_44753559224331
// MI455X (gfx1250) — matrix operand order repaired and checked by reading, not compiled, not run on hardware
//
#include <hip/hip_runtime.h>

// Problem constants (match reference)
#define NN   100000
#define NPAD 100032          // NN rounded up to multiple of 64 (GEMM M tile)
#define EE   800000
#define FIN  128
#define HID  512
#define NCLS 2
#define GEPS 1e-5f

typedef __attribute__((ext_vector_type(16))) __bf16 v16bf;
typedef __attribute__((ext_vector_type(8)))  float  v8f;

__device__ __forceinline__ unsigned short f2bf(float f) {
  union { float f; unsigned u; } v; v.f = f;
  unsigned r = v.u + 0x7FFFu + ((v.u >> 16) & 1u);   // round-to-nearest-even
  return (unsigned short)(r >> 16);
}
__device__ __forceinline__ __bf16 bfbits(unsigned short u) {
  return __builtin_bit_cast(__bf16, u);
}

// ---------------- elementwise fills / conversions ----------------
__global__ void k_fill(float* __restrict__ p, float v, size_t n) {
  size_t i = (size_t)blockIdx.x * blockDim.x + threadIdx.x;
  size_t s = (size_t)gridDim.x * blockDim.x;
  for (; i < n; i += s) p[i] = v;
}

__global__ void k_bias_rows(float* __restrict__ C, const float* __restrict__ b,
                            size_t total, int d) {
  size_t i = (size_t)blockIdx.x * blockDim.x + threadIdx.x;
  size_t s = (size_t)gridDim.x * blockDim.x;
  for (; i < total; i += s) C[i] = b[(int)(i % d)];
}

__global__ void k_f2bf_buf(const float* __restrict__ src, unsigned short* __restrict__ dst,
                           size_t n) {
  size_t i = (size_t)blockIdx.x * blockDim.x + threadIdx.x;
  size_t s = (size_t)gridDim.x * blockDim.x;
  for (; i < n; i += s) dst[i] = f2bf(src[i]);
}

// ---------------- gcn_norm ----------------
__global__ void k_deg(const int* __restrict__ ei, const float* __restrict__ w,
                      float* __restrict__ deg, int E_) {
  int e = blockIdx.x * blockDim.x + threadIdx.x;
  if (e < E_) atomicAdd(&deg[ei[E_ + e]], w[e]);     // deg over target (col)
}

__global__ void k_dinv(float* __restrict__ deg, int n) {
  int i = blockIdx.x * blockDim.x + threadIdx.x;
  if (i < n) { float d = deg[i]; deg[i] = d > 0.f ? rsqrtf(d) : 0.f; }
}

__global__ void k_norm(const int* __restrict__ ei, const float* __restrict__ w,
                       const float* __restrict__ dinv, float* __restrict__ nrm, int E_) {
  int e = blockIdx.x * blockDim.x + threadIdx.x;
  if (e < E_) nrm[e] = dinv[ei[e]] * w[e] * dinv[ei[E_ + e]];
}

// ---------------- SpMM (COO scatter-add) ----------------
__global__ void k_spmm(const float* __restrict__ src, float* __restrict__ dst,
                       const int* __restrict__ ei, const float* __restrict__ nrm,
                       int E_, int d) {
  int e = blockIdx.x;
  int r = ei[e], c = ei[E_ + e];
  float nv = nrm[e];
  const float* s = src + (size_t)r * d;
  float* p = dst + (size_t)c * d;
  for (int f = threadIdx.x; f < d; f += blockDim.x)
    atomicAdd(&p[f], s[f] * nv);
}

// ---------------- bf16 WMMA GEMM:  C[NPAD,Nd] += A[NPAD,Kd] @ Wbf[Kd,Nd] ----------------
// A is f32 (converted on the fly), Wbf is preconverted bf16 bits.
// Block = 256 threads = 8 waves (4 M-waves x 2 N-waves); block tile 64x128.
// Wave tile 16x64 = 4 accumulators; K step 32 => 4 WMMAs per wave per step.
// NO bounds checks: M padded to NPAD, Nd/Kd multiples of tile sizes.
__global__ __launch_bounds__(256) void k_gemm_wmma(
    const float* __restrict__ A, const unsigned short* __restrict__ Wbf,
    float* __restrict__ C, int Kd, int Nd)
{
  __shared__ __align__(16) unsigned short As[64][40];    // [m][k] (+8 pad)
  __shared__ __align__(16) unsigned short Ws[32][136];   // [k][n] (+8 pad)

  const int tid  = threadIdx.x;
  const int lane = tid & 31;
  const int wv   = tid >> 5;
  const int wm   = (wv >> 1) << 4;        // 0,16,32,48
  const int wn   = (wv & 1) << 6;         // 0,64
  const int m0   = blockIdx.x << 6;       // *64
  const int n0   = blockIdx.y << 7;       // *128

  v8f acc[4] = {};

  // per-thread staging coordinates
  const int am = (tid << 3) >> 5;         // A: 8 consecutive floats per thread
  const int ak = (tid << 3) & 31;
  const int wk = (tid << 4) >> 7;         // W: 16 consecutive ushorts per thread
  const int wnn = (tid << 4) & 127;

  for (int k0 = 0; k0 < Kd; k0 += 32) {
    // stage A tile 64x32 (f32 -> bf16), fully coalesced float4 pairs
    {
      const float* ap = A + (size_t)(m0 + am) * Kd + k0 + ak;
      float4 f0 = *(const float4*)ap;
      float4 f1 = *(const float4*)(ap + 4);
      As[am][ak + 0] = f2bf(f0.x); As[am][ak + 1] = f2bf(f0.y);
      As[am][ak + 2] = f2bf(f0.z); As[am][ak + 3] = f2bf(f0.w);
      As[am][ak + 4] = f2bf(f1.x); As[am][ak + 5] = f2bf(f1.y);
      As[am][ak + 6] = f2bf(f1.z); As[am][ak + 7] = f2bf(f1.w);
    }
    // stage W tile 32x128 (bf16 bits), two b128 loads + two b128 LDS stores
    {
      const unsigned short* wp = Wbf + (size_t)(k0 + wk) * Nd + n0 + wnn;
      uint4 w0 = *(const uint4*)wp;
      uint4 w1 = *(const uint4*)(wp + 8);
      *(uint4*)&Ws[wk][wnn + 0] = w0;
      *(uint4*)&Ws[wk][wnn + 8] = w1;
    }
    __syncthreads();

    // A fragment: lane<16 holds row M=lane%16, K in {0..7,16..23}; hi lanes +8
    v16bf af;
    const int rowA = wm + (lane & 15);
    const int hi   = (lane >> 4) << 3;     // 0 or 8
    #pragma unroll
    for (int j = 0; j < 16; ++j)
      af[j] = bfbits(As[rowA][j + (j & 8) + hi]);

    #pragma unroll
    for (int t = 0; t < 4; ++t) {
      v16bf bfrag;
      #pragma unroll
      for (int j = 0; j < 16; ++j)
        bfrag[j] = bfbits(Ws[j + (j & 8) + hi][wn + (t << 4) + (lane & 15)]);  // j = K row (same order as A), lane%16 = N col
      acc[t] = __builtin_amdgcn_wmma_f32_16x16x32_bf16(false, af, false, bfrag,
                                                       (short)0, acc[t], false, false);
    }
    __syncthreads();
  }

  // C layout: VGPR r -> lanes 0-15: M=r, N=lane; lanes 16-31: M=r+8, N=lane-16
  const int rbase = m0 + wm + ((lane >> 4) << 3);
  #pragma unroll
  for (int t = 0; t < 4; ++t) {
    const int col = n0 + wn + (t << 4) + (lane & 15);
    #pragma unroll
    for (int r = 0; r < 8; ++r) {
      size_t idx = (size_t)(rbase + r) * Nd + col;
      C[idx] += acc[t][r];
    }
  }
}

// ---------------- ELU (in place) + column sums ----------------
__global__ void k_elu_colsum(float* __restrict__ X, float* __restrict__ cs,
                             int n, int d, int rpb) {
  int r0 = blockIdx.x * rpb;
  int r1 = r0 + rpb; if (r1 > n) r1 = n;
  for (int f = threadIdx.x; f < d; f += blockDim.x) {
    float s = 0.f;
    for (int r = r0; r < r1; ++r) {
      size_t idx = (size_t)r * d + f;
      float v = X[idx];
      v = v > 0.f ? v : expm1f(v);
      X[idx] = v;
      s += v;
    }
    atomicAdd(&cs[f], s);
  }
}

// ---------------- center (x -= ms*mean) + variance partials ----------------
__global__ void k_center_var(float* __restrict__ X, const float* __restrict__ cs,
                             const float* __restrict__ ms, float* __restrict__ var,
                             float invN, int n, int d, int rpb) {
  int r0 = blockIdx.x * rpb;
  int r1 = r0 + rpb; if (r1 > n) r1 = n;
  for (int f = threadIdx.x; f < d; f += blockDim.x) {
    float sub = ms[f] * cs[f] * invN;
    float s = 0.f;
    for (int r = r0; r < r1; ++r) {
      size_t idx = (size_t)r * d + f;
      float v = X[idx] - sub;
      X[idx] = v;
      s += v * v;
    }
    atomicAdd(&var[f], s);
  }
}

__global__ void k_gnorm_fin(float* __restrict__ X, const float* __restrict__ var,
                            const float* __restrict__ w, const float* __restrict__ b,
                            float invN, size_t total, int d) {
  size_t i = (size_t)blockIdx.x * blockDim.x + threadIdx.x;
  size_t s = (size_t)gridDim.x * blockDim.x;
  for (; i < total; i += s) {
    int f = (int)(i % d);
    X[i] = X[i] * rsqrtf(var[f] * invN + GEPS) * w[f] + b[f];
  }
}

// ---------------- final projection to C=2 (wave per node) ----------------
__global__ void k_gemv2_acc(const float* __restrict__ h, const float* __restrict__ Wk,
                            float* __restrict__ out, int d, int n) {
  int gid  = blockIdx.x * blockDim.x + threadIdx.x;
  int node = gid >> 5, lane = gid & 31;
  if (node >= n) return;
  const float* hr = h + (size_t)node * d;
  float s0 = 0.f, s1 = 0.f;
  for (int k = lane; k < d; k += 32) {
    float hv = hr[k];
    s0 += hv * Wk[2 * k];
    s1 += hv * Wk[2 * k + 1];
  }
  for (int off = 16; off > 0; off >>= 1) {
    s0 += __shfl_down(s0, off, 32);
    s1 += __shfl_down(s1, off, 32);
  }
  if (lane == 0) { out[2 * node] += s0; out[2 * node + 1] += s1; }
}

// ---------------- host orchestration ----------------
extern "C" void kernel_launch(void* const* d_in, const int* in_sizes, int n_in,
                              void* d_out, int out_size, void* d_ws, size_t ws_size,
                              hipStream_t stream) {
  (void)in_sizes; (void)n_in; (void)out_size; (void)ws_size;

  const float* x     = (const float*)d_in[0];
  const float* wgt   = (const float*)d_in[1];
  const float* W1    = (const float*)d_in[2];
  const float* b1    = (const float*)d_in[3];
  const float* W2    = (const float*)d_in[4];
  const float* b2    = (const float*)d_in[5];
  const float* W3    = (const float*)d_in[6];
  const float* b3    = (const float*)d_in[7];
  const float* gn1w  = (const float*)d_in[8];
  const float* gn1b  = (const float*)d_in[9];
  const float* gn1ms = (const float*)d_in[10];
  const float* gn2w  = (const float*)d_in[11];
  const float* gn2b  = (const float*)d_in[12];
  const float* gn2ms = (const float*)d_in[13];
  const int*   ei    = (const int*)d_in[14];     // [2,E] (row, col)
  float* out = (float*)d_out;

  // workspace carve-up
  char* ws = (char*)d_ws;
  size_t off = 0;
  auto carve = [&](size_t bytes) -> void* {
    void* p = (void*)(ws + off);
    off += (bytes + 255) & ~(size_t)255;
    return p;
  };
  float* deg  = (float*)carve((size_t)NN * 4);           // then holds dinv
  float* nrm  = (float*)carve((size_t)EE * 4);
  float* mean = (float*)carve((size_t)HID * 4);
  float* var  = (float*)carve((size_t)HID * 4);
  unsigned short* W1bf = (unsigned short*)carve((size_t)4 * FIN * HID * 2);
  unsigned short* W2bf = (unsigned short*)carve((size_t)4 * HID * HID * 2);
  float* A = (float*)carve((size_t)NPAD * HID * 4);
  float* B = (float*)carve((size_t)NPAD * HID * 4);
  float* O = (float*)carve((size_t)NPAD * HID * 4);

  const float invN = 1.0f / (float)NN;

  auto fill = [&](float* p, float v, size_t n) {
    k_fill<<<2048, 256, 0, stream>>>(p, v, n);
  };
  auto gemm = [&](const float* Ain, const unsigned short* Wk, float* Cc, int Kd) {
    dim3 g(NPAD / 64, HID / 128);
    k_gemm_wmma<<<g, 256, 0, stream>>>(Ain, Wk, Cc, Kd, HID);
  };
  auto spmm = [&](const float* s, float* dptr, int d) {
    fill(dptr, 0.f, (size_t)NN * d);
    k_spmm<<<EE, (d < 256 ? d : 256), 0, stream>>>(s, dptr, ei, nrm, EE, d);
  };
  auto elu_gnorm = [&](float* X, const float* gw, const float* gb, const float* gms) {
    fill(mean, 0.f, HID);
    k_elu_colsum<<<(NN + 127) / 128, 256, 0, stream>>>(X, mean, NN, HID, 128);
    fill(var, 0.f, HID);
    k_center_var<<<(NN + 127) / 128, 256, 0, stream>>>(X, mean, gms, var, invN, NN, HID, 128);
    k_gnorm_fin<<<2048, 256, 0, stream>>>(X, var, gw, gb, invN, (size_t)NN * HID, HID);
  };

  // ---- preconvert weights to bf16 (once per launch) ----
  k_f2bf_buf<<<512, 256, 0, stream>>>(W1, W1bf, (size_t)4 * FIN * HID);
  k_f2bf_buf<<<2048, 256, 0, stream>>>(W2, W2bf, (size_t)4 * HID * HID);

  // ---- gcn_norm ----
  fill(deg, 0.f, NN);
  k_deg<<<(EE + 255) / 256, 256, 0, stream>>>(ei, wgt, deg, EE);
  k_dinv<<<(NN + 255) / 256, 256, 0, stream>>>(deg, NN);
  k_norm<<<(EE + 255) / 256, 256, 0, stream>>>(ei, wgt, deg, nrm, EE);

  // ---- layer 1: TAGConv(128 -> 512) + ELU + GraphNorm ----
  hipMemcpyAsync(A, x, (size_t)NN * FIN * 4, hipMemcpyDeviceToDevice, stream);
  k_bias_rows<<<2048, 256, 0, stream>>>(O, b1, (size_t)NN * HID, HID);
  gemm(A, W1bf + 0 * (size_t)FIN * HID, O, FIN);           // hop0 = x (staged)
  spmm(A, B, FIN);  gemm(B, W1bf + 1 * (size_t)FIN * HID, O, FIN);
  spmm(B, A, FIN);  gemm(A, W1bf + 2 * (size_t)FIN * HID, O, FIN);
  spmm(A, B, FIN);  gemm(B, W1bf + 3 * (size_t)FIN * HID, O, FIN);
  elu_gnorm(O, gn1w, gn1b, gn1ms);

  // ---- layer 2: TAGConv(512 -> 512) + ELU + GraphNorm ----
  hipMemcpyAsync(A, O, (size_t)NN * HID * 4, hipMemcpyDeviceToDevice, stream);
  k_bias_rows<<<2048, 256, 0, stream>>>(O, b2, (size_t)NN * HID, HID);
  gemm(A, W2bf + 0 * (size_t)HID * HID, O, HID);           // hop0
  spmm(A, B, HID);  gemm(B, W2bf + 1 * (size_t)HID * HID, O, HID);
  spmm(B, A, HID);  gemm(A, W2bf + 2 * (size_t)HID * HID, O, HID);
  spmm(A, B, HID);  gemm(B, W2bf + 3 * (size_t)HID * HID, O, HID);
  elu_gnorm(O, gn2w, gn2b, gn2ms);

  // ---- layer 3: TAGConv(512 -> 2) ----
  hipMemcpyAsync(A, O, (size_t)NN * HID * 4, hipMemcpyDeviceToDevice, stream);
  k_bias_rows<<<2048, 256, 0, stream>>>(out, b3, (size_t)NN * NCLS, NCLS);
  auto gemv = [&](const float* h, int k) {
    k_gemv2_acc<<<((size_t)NN * 32 + 255) / 256, 256, 0, stream>>>(
        h, W3 + (size_t)k * HID * NCLS, out, HID, NN);
  };
  gemv(A, 0);
  spmm(A, B, HID); gemv(B, 1);
  spmm(B, A, HID); gemv(A, 2);
  spmm(A, B, HID); gemv(B, 3);
}
